// GSC_Vanilla_SNN_86474871538256
// MI455X (gfx1250) — compile-verified, matrix operand order checked by reading, not run on hardware
//
#include <hip/hip_runtime.h>

// ---------------------------------------------------------------------------
// Fused SNN (3x LIF + readout, S=101 steps) for gfx1250 using f16 WMMA with
// f32 accumulation. One block owns BT batch rows for the entire time loop;
// membrane potentials live in VGPRs, spikes flow between waves via LDS.
// Weights are pre-transposed/converted to f16 in d_ws by a prep kernel and
// streamed from L2 as WMMA B-fragments.
// ---------------------------------------------------------------------------

typedef __attribute__((ext_vector_type(16))) _Float16 v16h;
typedef __attribute__((ext_vector_type(8)))  _Float16 v8h;
typedef __attribute__((ext_vector_type(8)))  float    v8f;

#define SNN_B     2048
#define SNN_C     3
#define SNN_S     101
#define SNN_M     40
#define SNN_DIN   120     // C*M
#define SNN_H     200
#define SNN_DOUT  12
#define KP1       128     // padded K for layer 1 (4 k-tiles of 32)
#define HP        208     // padded hidden N (13 n-tiles of 16)
#define KP2       224     // padded K over H (7 k-tiles of 32)
#define BT        32      // batch rows per block
#define NTILES    26      // (BT/16) * (HP/16) output tiles per hidden GEMM
#define MAXSLOTS  4       // ceil(NTILES / 8 waves)

// f16 A/B fragment loader. For v_wmma_f32_16x16x32_f16 the 16-bit operand
// layout is: lanes 0-15 hold K {kb..kb+7, kb+16..kb+23}, lanes 16-31 hold
// K {kb+8..kb+15, kb+24..kb+31} of row (m or n) = lane&15.  The caller passes
// kOfs = kb + (lane>=16 ? 8 : 0); we fetch two 16B chunks 32B apart.
__device__ __forceinline__ v16h ld_frag(const _Float16* buf, int row, int kOfs, int stride) {
  const _Float16* p = buf + (size_t)row * stride + kOfs;
  v8h c0 = *(const v8h*)(p);
  v8h c1 = *(const v8h*)(p + 16);
  return __builtin_shufflevector(c0, c1, 0, 1, 2, 3, 4, 5, 6, 7,
                                         8, 9, 10, 11, 12, 13, 14, 15);
}

// One hidden layer: tiled GEMM (A f16 from LDS, Wt f16 [N][K] from L2) with
// f32 WMMA accumulation, + bias, LIF update (tau=2, Vth=1, hard reset to 0),
// spike written back to LDS as f16 for the next layer's A matrix.
template <int KT>
__device__ __forceinline__ void gemm_lif(
    const _Float16* A, int As,
    const _Float16* __restrict__ Wt, int Ws,
    const float* __restrict__ bias, int nBias,
    _Float16* sOut, int So,
    v8f (&vst)[MAXSLOTS],
    int wave, int ln, int lhi)
{
#pragma unroll
  for (int s = 0; s < MAXSLOTS; ++s) {
    int tile = wave + s * 8;
    if (tile < NTILES) {
      int rowBase = (tile & 1) * 16;   // BT/16 == 2 row tiles
      int nBase   = (tile >> 1) * 16;  // 13 column tiles
      v8f acc = {0.f, 0.f, 0.f, 0.f, 0.f, 0.f, 0.f, 0.f};
#pragma unroll
      for (int kt = 0; kt < KT; ++kt) {
        v16h a = ld_frag(A,  rowBase + ln, kt * 32 + lhi * 8, As);
        v16h b = ld_frag(Wt, nBase   + ln, kt * 32 + lhi * 8, Ws);
        acc = __builtin_amdgcn_wmma_f32_16x16x32_f16(
            false, a, false, b, (short)0, acc, false, false);
      }
      int n = nBase + ln;                 // this lane's output column
      float bv = (n < nBias) ? bias[n] : 0.0f;
      v8f v = vst[s];
      _Float16* so = sOut + (size_t)(rowBase + lhi * 8) * So + n;
#pragma unroll
      for (int r = 0; r < 8; ++r) {       // C/D layout: vgpr r -> row r (+8 hi)
        float cur = acc[r] + bv;
        float vv  = 0.5f * (v[r] + cur);          // v += (cur - v)/tau, tau=2
        float sp  = (vv >= 1.0f) ? 1.0f : 0.0f;   // Heaviside(v - Vth)
        v[r] = vv * (1.0f - sp);                  // hard reset
        so[(size_t)r * So] = (_Float16)sp;        // exact in f16
      }
      vst[s] = v;
    }
  }
}

__global__ __launch_bounds__(256) void snn_fused(
    const float* __restrict__ x,
    const float* __restrict__ b1, const float* __restrict__ b2,
    const float* __restrict__ b3, const float* __restrict__ br,
    const _Float16* __restrict__ W1t, const _Float16* __restrict__ W2t,
    const _Float16* __restrict__ W3t, const _Float16* __restrict__ Wrt,
    float* __restrict__ out)
{
  __shared__ __align__(16) _Float16 xA[BT * KP1];  // layer-1 A (f16)
  __shared__ __align__(16) _Float16 sA[BT * KP2];  // s1 / s3 (shared, see barriers)
  __shared__ __align__(16) _Float16 sB[BT * KP2];  // s2
  __shared__ __align__(16) float    outBuf[BT * 16];

  const int tid   = threadIdx.x;
  const int wave  = tid >> 5;     // 8 waves of 32 (wave32)
  const int lane  = tid & 31;
  const int ln    = lane & 15;
  const int lhi   = lane >> 4;
  const int bBase = blockIdx.x * BT;

  // Zero-init LDS once: padded K columns [200,224) / [120,128) must stay 0.
  for (int i = tid; i < BT * KP2; i += 256) { sA[i] = (_Float16)0.f; sB[i] = (_Float16)0.f; }
  for (int i = tid; i < BT * KP1; i += 256) xA[i] = (_Float16)0.f;

  const v8f vz = {0.f, 0.f, 0.f, 0.f, 0.f, 0.f, 0.f, 0.f};
  v8f v1s[MAXSLOTS], v2s[MAXSLOTS], v3s[MAXSLOTS];
#pragma unroll
  for (int s = 0; s < MAXSLOTS; ++s) { v1s[s] = vz; v2s[s] = vz; v3s[s] = vz; }
  v8f outAcc = vz;   // readout accumulator over time (waves 0,1 only)

  __syncthreads();

  for (int t = 0; t < SNN_S; ++t) {
    // Stage 1: gather x[:, :, t, :] -> f16 row-major [row][k] in LDS.
    for (int i = tid; i < BT * SNN_DIN; i += 256) {
      int row = i / SNN_DIN;
      int k   = i - row * SNN_DIN;
      int c   = k / SNN_M;
      int m   = k - c * SNN_M;
      size_t gi = (((size_t)(bBase + row) * SNN_C + c) * SNN_S + t) * SNN_M + m;
      xA[row * KP1 + k] = (_Float16)x[gi];
    }
    __syncthreads();

    // Layer 1: x @ W1  (K = 128 padded)
    gemm_lif<4>(xA, KP1, W1t, KP1, b1, SNN_H, sA, KP2, v1s, wave, ln, lhi);
    __syncthreads();
    // Layer 2: s1 @ W2 (K = 224 padded)
    gemm_lif<7>(sA, KP2, W2t, KP2, b2, SNN_H, sB, KP2, v2s, wave, ln, lhi);
    __syncthreads();
    // Layer 3: s2 @ W3
    gemm_lif<7>(sB, KP2, W3t, KP2, b3, SNN_H, sA, KP2, v3s, wave, ln, lhi);
    __syncthreads();

    // Readout: s3 @ Wr + br, accumulated over time in the WMMA C operand.
    // Waves 0/1 own the two 16-row tiles. Safe vs next-step sA writes: those
    // happen only after the stage-1 barrier, which these waves reach after
    // finishing their reads here.
    if (wave < 2) {
      v8f acc = outAcc;
#pragma unroll
      for (int kt = 0; kt < 7; ++kt) {
        v16h a = ld_frag(sA,  wave * 16 + ln, kt * 32 + lhi * 8, KP2);
        v16h b = ld_frag(Wrt, ln,             kt * 32 + lhi * 8, KP2);
        acc = __builtin_amdgcn_wmma_f32_16x16x32_f16(
            false, a, false, b, (short)0, acc, false, false);
      }
      float bb = (ln < SNN_DOUT) ? br[ln] : 0.0f;
#pragma unroll
      for (int r = 0; r < 8; ++r) acc[r] += bb;
      outAcc = acc;
    }
  }

  // Mean over time -> LDS, then per-row log_softmax over the 12 classes.
  if (wave < 2) {
#pragma unroll
    for (int r = 0; r < 8; ++r) {
      int row = wave * 16 + lhi * 8 + r;
      outBuf[row * 16 + ln] = outAcc[r] * (1.0f / (float)SNN_S);
    }
  }
  __syncthreads();
  if (tid < BT) {
    float vals[SNN_DOUT];
    float mx = -3.4e38f;
#pragma unroll
    for (int j = 0; j < SNN_DOUT; ++j) {
      vals[j] = outBuf[tid * 16 + j];
      mx = fmaxf(mx, vals[j]);
    }
    float sum = 0.0f;
#pragma unroll
    for (int j = 0; j < SNN_DOUT; ++j) sum += __expf(vals[j] - mx);
    float lse = __logf(sum);
#pragma unroll
    for (int j = 0; j < SNN_DOUT; ++j)
      out[(size_t)(bBase + tid) * SNN_DOUT + j] = vals[j] - mx - lse;
  }
}

// ---------------------------------------------------------------------------
// Prep: convert fp32 weights [K][N] to f16, transposed [N][K] with zero
// padding, into d_ws. Runs every launch (deterministic, ~250 KB total).
// ---------------------------------------------------------------------------
__global__ __launch_bounds__(256) void snn_prep(
    const float* __restrict__ W1, const float* __restrict__ W2,
    const float* __restrict__ W3, const float* __restrict__ Wr,
    _Float16* __restrict__ W1t, _Float16* __restrict__ W2t,
    _Float16* __restrict__ W3t, _Float16* __restrict__ Wrt)
{
  int i = blockIdx.x * blockDim.x + threadIdx.x;
  if (i < HP * KP1) {
    int n = i / KP1, k = i - n * KP1;
    W1t[i] = (n < SNN_H && k < SNN_DIN) ? (_Float16)W1[k * SNN_H + n] : (_Float16)0.f;
  }
  if (i < HP * KP2) {
    int n = i / KP2, k = i - n * KP2;
    bool ok = (n < SNN_H && k < SNN_H);
    W2t[i] = ok ? (_Float16)W2[k * SNN_H + n] : (_Float16)0.f;
    W3t[i] = ok ? (_Float16)W3[k * SNN_H + n] : (_Float16)0.f;
  }
  if (i < 16 * KP2) {
    int n = i / KP2, k = i - n * KP2;
    Wrt[i] = (n < SNN_DOUT && k < SNN_H) ? (_Float16)Wr[k * SNN_DOUT + n] : (_Float16)0.f;
  }
}

extern "C" void kernel_launch(void* const* d_in, const int* in_sizes, int n_in,
                              void* d_out, int out_size, void* d_ws, size_t ws_size,
                              hipStream_t stream) {
  const float* x  = (const float*)d_in[0];
  const float* W1 = (const float*)d_in[1];
  const float* b1 = (const float*)d_in[2];
  const float* W2 = (const float*)d_in[3];
  const float* b2 = (const float*)d_in[4];
  const float* W3 = (const float*)d_in[5];
  const float* b3 = (const float*)d_in[6];
  const float* Wr = (const float*)d_in[7];
  const float* br = (const float*)d_in[8];

  _Float16* W1t = (_Float16*)d_ws;              // [208][128]
  _Float16* W2t = W1t + HP * KP1;               // [208][224]
  _Float16* W3t = W2t + HP * KP2;               // [208][224]
  _Float16* Wrt = W3t + HP * KP2;               // [16][224]

  int prepElems = HP * KP2;                     // largest region: 46592
  snn_prep<<<(prepElems + 255) / 256, 256, 0, stream>>>(W1, W2, W3, Wr,
                                                        W1t, W2t, W3t, Wrt);
  snn_fused<<<SNN_B / BT, 256, 0, stream>>>(x, b1, b2, b3, br,
                                            W1t, W2t, W3t, Wrt,
                                            (float*)d_out);
}
